// Discriminator_81862076662045
// MI455X (gfx1250) — matrix operand order repaired and checked by reading, not compiled, not run on hardware
//
#include <hip/hip_runtime.h>
#include <hip/hip_bf16.h>

// ---------------------------------------------------------------------------
// MI455X (gfx1250, wave32) implementation.
// All dense contractions use v_wmma_f32_16x16x32_bf16 (bf16 operands, f32 acc).
// GEMM stages tiles via CDNA5 async global->LDS copies (ASYNCcnt) with double
// buffering when the toolchain exposes the builtin.
// ---------------------------------------------------------------------------

typedef __attribute__((ext_vector_type(16))) __bf16 v16bf;
typedef __attribute__((ext_vector_type(8)))  float  v8f;
typedef __bf16 bf16_t;

namespace {
constexpr int NB = 8, NI = 64, NS = 64, NH = 256, NL = 3;
constexpr int NM = NB * NI * NS;          // 32768 rows in every [M,H] activation
}

#ifndef __has_builtin
#define __has_builtin(x) 0
#endif
#if __has_builtin(__builtin_amdgcn_global_load_async_to_lds_b128)
#define ASYNC_LDS 1
#endif

// Parameter types of the async builtin: v4i in AS1 (global) / AS3 (LDS).
typedef int v4i __attribute__((vector_size(16)));
typedef __attribute__((address_space(1))) v4i* gptr_v4i;
typedef __attribute__((address_space(3))) v4i* lptr_v4i;

// 16-byte global->LDS copy (async on CDNA5, sync fallback otherwise)
__device__ __forceinline__ void cp16(const bf16_t* __restrict__ g, bf16_t* l) {
#ifdef ASYNC_LDS
    __builtin_amdgcn_global_load_async_to_lds_b128(
        (gptr_v4i)(size_t)g, (lptr_v4i)(unsigned)(size_t)l, 0, 0);
#else
    *(uint4*)l = *(const uint4*)g;
#endif
}

__device__ __forceinline__ void async_wait() {
#ifdef ASYNC_LDS
#if __has_builtin(__builtin_amdgcn_s_wait_asynccnt)
    __builtin_amdgcn_s_wait_asynccnt(0);
#else
    asm volatile("s_wait_asynccnt 0x0" ::: "memory");
#endif
#endif
}

// ---- fp32 -> bf16 convert (weights) ---------------------------------------
__global__ void k_f32_to_bf16(const float* __restrict__ in, bf16_t* __restrict__ out, int n) {
    int i = blockIdx.x * blockDim.x + threadIdx.x;
    if (i < n) out[i] = (bf16_t)in[i];
}

// ---- 1x1 conv expand: hl = x * conv_w + conv_b ----------------------------
__global__ void k_conv(const float* __restrict__ x, const float* __restrict__ cw,
                       const float* __restrict__ cb, float* __restrict__ hf,
                       bf16_t* __restrict__ hb) {
    int idx = blockIdx.x * 256 + threadIdx.x;      // NM*NH total
    int h = idx & (NH - 1);
    int p = idx >> 8;                               // (b,i,s) flat
    float v = x[p] * cw[h] + cb[h];
    hf[idx] = v;
    hb[idx] = (bf16_t)v;
}

// ---- ES[i,s,h] = sum_j adj[i,j] * was[j,s,h] ------------------------------
__global__ void k_es(const float* __restrict__ adj, const float* __restrict__ was,
                     float* __restrict__ es) {
    int idx = blockIdx.x * 256 + threadIdx.x;      // NI*NS*NH total
    int h = idx & (NH - 1);
    int s = (idx >> 8) & (NS - 1);
    int i = idx >> 14;
    const float* arow = adj + i * NI;
    float acc = 0.f;
    #pragma unroll 4
    for (int j = 0; j < NI; ++j)
        acc += arow[j] * was[((size_t)(j * NS + s)) * NH + h];
    es[idx] = acc;
}

// ---- out = bf16(a + b[idx & mask])  (XS = hl+ES, XT = hl+pos) -------------
__global__ void k_add_bf16(const float* __restrict__ a, const float* __restrict__ b,
                           bf16_t* __restrict__ out, unsigned mask) {
    int idx = blockIdx.x * 256 + threadIdx.x;
    out[idx] = (bf16_t)(a[idx] + b[idx & mask]);
}

// ---------------------------------------------------------------------------
// Generic GEMM: C[M,N] = A[M,K](bf16) * W[N,K]^T(bf16, row stride ldw)
//  + bias (opt) + Cin (opt accumulate) ; optional ReLU ; writes fp32 and/or
//  bf16 outputs.  Block tile 128x64, 8 waves (4x2), each wave 2x2 WMMA tiles.
//  Double-buffered async global->LDS pipeline, one barrier per k-step.
// ---------------------------------------------------------------------------
__global__ __launch_bounds__(256)
void k_gemm(const bf16_t* __restrict__ A, const bf16_t* __restrict__ Wt,
            const float* __restrict__ bias, const float* __restrict__ Cin,
            float* __restrict__ Cf, bf16_t* __restrict__ Cb,
            int Nn, int Kk, int ldw, int relu) {
    __shared__ bf16_t As[2][128][32];   // 2 x 8KB
    __shared__ bf16_t Bs[2][64][32];    // 2 x 4KB
    const int tid  = threadIdx.x;
    const int lane = tid & 31, wid = tid >> 5;
    const int wm = wid & 3, wn = wid >> 2;          // 4 x 2 wave grid
    const int m0 = blockIdx.x * 128, n0 = blockIdx.y * 64;
    const int ml = lane & 15, hf = lane >> 4;

    // Per-thread copy slots: A tile 128x32 = 512 uint4 (2/thread),
    // W tile 64x32 = 256 uint4 (1/thread).
    const int crow = tid >> 2;            // 0..63
    const int segE = (tid & 3) * 8;       // bf16 element offset within row
    const bf16_t* gA0 = A + (size_t)(m0 + crow) * Kk + segE;
    const bf16_t* gA1 = A + (size_t)(m0 + 64 + crow) * Kk + segE;
    const bf16_t* gB  = Wt + (size_t)(n0 + crow) * ldw + segE;

    auto issue = [&](int buf, int k0) {
        cp16(gA0 + k0, &As[buf][crow][segE]);
        cp16(gA1 + k0, &As[buf][64 + crow][segE]);
        cp16(gB  + k0, &Bs[buf][crow][segE]);
    };

    v8f acc[2][2];
    for (int a = 0; a < 2; ++a)
        for (int b = 0; b < 2; ++b)
            for (int r = 0; r < 8; ++r) acc[a][b][r] = 0.f;

    const int nsteps = Kk >> 5;
    issue(0, 0);                                   // prologue: fill buffer 0

    for (int step = 0; step < nsteps; ++step) {
        const int buf = step & 1;
        async_wait();                              // my fills of As/Bs[buf] done
        __syncthreads();                           // everyone's fills visible;
                                                   // everyone done reading buf^1
        if (step + 1 < nsteps) issue(buf ^ 1, (step + 1) << 5);

        v16bf af[2], bfr[2];
        #pragma unroll
        for (int mt = 0; mt < 2; ++mt) {
            int arow = wm * 32 + mt * 16 + ml;
            #pragma unroll
            for (int e = 0; e < 16; ++e) {          // A 16x32 bf16 lane layout
                int kl = hf * 16 + e;
                af[mt][e] = As[buf][arow][kl];
            }
        }
        #pragma unroll
        for (int nt = 0; nt < 2; ++nt) {
            int brow = wn * 32 + nt * 16 + ml;      // B[k][n] = W[n][k]
            #pragma unroll
            for (int e = 0; e < 16; ++e) {
                int kl = hf * 16 + e;
                bfr[nt][e] = Bs[buf][brow][kl];
            }
        }
        #pragma unroll
        for (int mt = 0; mt < 2; ++mt)
            #pragma unroll
            for (int nt = 0; nt < 2; ++nt)
                acc[mt][nt] = __builtin_amdgcn_wmma_f32_16x16x32_bf16(
                    false, af[mt], false, bfr[nt], (short)0, acc[mt][nt],
                    false, false);
    }

    // Epilogue: D 16x16 f32 layout (lane n = lane&15, m = r + (lane>>4)*8)
    for (int mt = 0; mt < 2; ++mt)
        for (int nt = 0; nt < 2; ++nt) {
            int n = n0 + wn * 32 + nt * 16 + ml;
            float bv = bias ? bias[n] : 0.f;
            for (int r = 0; r < 8; ++r) {
                int m = m0 + wm * 32 + mt * 16 + hf * 8 + r;
                size_t o = (size_t)m * Nn + n;
                float v = acc[mt][nt][r] + bv;
                if (Cin)  v += Cin[o];
                if (relu) v = fmaxf(v, 0.f);
                if (Cf) Cf[o] = v;
                if (Cb) Cb[o] = (bf16_t)v;
            }
        }
}

// ---------------------------------------------------------------------------
// energy[b,h,i,j] = (1/sqrt(H)) * sum_s q[b,i,s,h] * k[b,j,s,h]
// One wave per 16x16 (i,j) tile; K-dim = s (two steps of 32).
// ---------------------------------------------------------------------------
__global__ __launch_bounds__(256)
void k_energy(const bf16_t* __restrict__ q, const bf16_t* __restrict__ kk,
              float* __restrict__ att, float scale) {
    const int lane = threadIdx.x & 31;
    const int t = blockIdx.x * 8 + (threadIdx.x >> 5);
    const int tj = t & 3, ti = (t >> 2) & 3, h = (t >> 4) & (NH - 1), b = t >> 12;
    const int ml = lane & 15, hf = lane >> 4;
    v8f acc; for (int r = 0; r < 8; ++r) acc[r] = 0.f;

    for (int s0 = 0; s0 < NS; s0 += 32) {
        v16bf af, bfr;
        size_t abase = ((size_t)(b * NI + ti * 16 + ml) * NS) * NH + h;
        #pragma unroll
        for (int e = 0; e < 16; ++e) {
            int s = s0 + hf * 16 + e;
            af[e] = q[abase + (size_t)s * NH];
        }
        size_t bbase = ((size_t)(b * NI + tj * 16 + ml) * NS) * NH + h;
        #pragma unroll
        for (int e = 0; e < 16; ++e) {
            int s = s0 + hf * 16 + e;
            bfr[e] = kk[bbase + (size_t)s * NH];
        }
        acc = __builtin_amdgcn_wmma_f32_16x16x32_bf16(
            false, af, false, bfr, (short)0, acc, false, false);
    }
    for (int r = 0; r < 8; ++r) {
        int i = ti * 16 + hf * 8 + r, j = tj * 16 + ml;
        att[((size_t)(b * NH + h) * NI + i) * NI + j] = acc[r] * scale;
    }
}

// ---- softmax over j (last, contiguous dim of att[b,h,i,:]) ----------------
__global__ void k_softmax(float* __restrict__ att) {
    int row = blockIdx.x * 256 + threadIdx.x;      // NB*NH*NI rows
    float* p = att + (size_t)row * NI;
    float vals[NI];
    float mx = -3.0e38f;
    #pragma unroll
    for (int j = 0; j < NI; ++j) { vals[j] = p[j]; mx = fmaxf(mx, vals[j]); }
    float sum = 0.f;
    #pragma unroll
    for (int j = 0; j < NI; ++j) { vals[j] = expf(vals[j] - mx); sum += vals[j]; }
    float inv = 1.f / sum;
    #pragma unroll
    for (int j = 0; j < NI; ++j) p[j] = vals[j] * inv;
}

// ---------------------------------------------------------------------------
// ctx[b,i,s,h] = sum_j att[b,h,i,j] * v[b,j,s,h]   (one wave per 16x16 tile)
// ---------------------------------------------------------------------------
__global__ __launch_bounds__(256)
void k_context(const float* __restrict__ att, const bf16_t* __restrict__ v,
               bf16_t* __restrict__ ctx) {
    const int lane = threadIdx.x & 31;
    const int t = blockIdx.x * 8 + (threadIdx.x >> 5);
    const int ts = t & 3, ti = (t >> 2) & 3, h = (t >> 4) & (NH - 1), b = t >> 12;
    const int ml = lane & 15, hf = lane >> 4;
    v8f acc; for (int r = 0; r < 8; ++r) acc[r] = 0.f;

    for (int j0 = 0; j0 < NI; j0 += 32) {
        v16bf af, bfr;
        size_t abase = ((size_t)(b * NH + h) * NI + (ti * 16 + ml)) * NI;
        #pragma unroll
        for (int e = 0; e < 16; ++e) {
            int j = j0 + hf * 16 + e;
            af[e] = (bf16_t)att[abase + j];
        }
        #pragma unroll
        for (int e = 0; e < 16; ++e) {
            int j = j0 + hf * 16 + e;
            bfr[e] = v[((size_t)(b * NI + j) * NS + ts * 16 + ml) * NH + h];
        }
        acc = __builtin_amdgcn_wmma_f32_16x16x32_bf16(
            false, af, false, bfr, (short)0, acc, false, false);
    }
    for (int r = 0; r < 8; ++r) {
        int i = ti * 16 + hf * 8 + r, s = ts * 16 + ml;
        ctx[((size_t)(b * NI + i) * NS + s) * NH + h] = (bf16_t)acc[r];
    }
}

// ---- LayerNorm over H of (ff + res); one wave per row; bf16 output --------
__global__ void k_layernorm(const float* __restrict__ ff, const float* __restrict__ res,
                            const float* __restrict__ g, const float* __restrict__ be,
                            bf16_t* __restrict__ out) {
    const int lane = threadIdx.x & 31;
    const int row  = blockIdx.x * 8 + (threadIdx.x >> 5);
    const float* pf = ff + (size_t)row * NH;
    const float* pr = res + (size_t)row * NH;
    float x[8], sum = 0.f, sq = 0.f;
    #pragma unroll
    for (int t = 0; t < 8; ++t) {
        int c = lane + t * 32;
        x[t] = pf[c] + pr[c];
        sum += x[t]; sq += x[t] * x[t];
    }
    for (int off = 16; off > 0; off >>= 1) {
        sum += __shfl_xor(sum, off, 32);
        sq  += __shfl_xor(sq,  off, 32);
    }
    float mean = sum * (1.f / NH);
    float var  = sq * (1.f / NH) - mean * mean;
    float inv  = rsqrtf(var + 1e-5f);
    #pragma unroll
    for (int t = 0; t < 8; ++t) {
        int c = lane + t * 32;
        out[(size_t)row * NH + c] = (bf16_t)((x[t] - mean) * inv * g[c] + be[c]);
    }
}

// ---- final dot: deterministic two-stage reduction + sigmoid ---------------
__global__ void k_dot_partial(const float* __restrict__ h, const float* __restrict__ w,
                              float* __restrict__ part) {
    __shared__ float red[256];
    const int bx = blockIdx.x;                 // b*128 + chunk
    const int b = bx >> 7, chunk = bx & 127;
    const size_t hbase = (size_t)b * (NI * NS * NH) + (size_t)chunk * 8192;
    const size_t wbase = (size_t)chunk * 8192;
    const int t = threadIdx.x;
    float s = 0.f;
    for (int i = 0; i < 32; ++i) {
        int o = t + i * 256;
        s += h[hbase + o] * w[wbase + o];
    }
    red[t] = s; __syncthreads();
    for (int off = 128; off > 0; off >>= 1) {
        if (t < off) red[t] += red[t + off];
        __syncthreads();
    }
    if (t == 0) part[bx] = red[0];
}

__global__ void k_dot_final(const float* __restrict__ part, const float* __restrict__ b1,
                            float* __restrict__ out) {
    __shared__ float red[128];
    const int b = blockIdx.x, t = threadIdx.x;   // 128 threads
    red[t] = part[b * 128 + t]; __syncthreads();
    for (int off = 64; off > 0; off >>= 1) {
        if (t < off) red[t] += red[t + off];
        __syncthreads();
    }
    if (t == 0) out[b] = 1.f / (1.f + expf(-(red[0] + b1[0])));
}

// ---------------------------------------------------------------------------
extern "C" void kernel_launch(void* const* d_in, const int* in_sizes, int n_in,
                              void* d_out, int out_size, void* d_ws, size_t ws_size,
                              hipStream_t stream) {
    const float* x      = (const float*)d_in[0];
    const float* adj    = (const float*)d_in[1];
    const float* cw     = (const float*)d_in[2];
    const float* cb     = (const float*)d_in[3];
    const float* sp_w   = (const float*)d_in[4];
    const float* sp_b   = (const float*)d_in[5];
    const float* sp_was = (const float*)d_in[6];
    const float* sp_g   = (const float*)d_in[7];
    const float* sp_be  = (const float*)d_in[8];
    const float* tp_w   = (const float*)d_in[9];
    const float* tp_b   = (const float*)d_in[10];
    const float* tp_pos = (const float*)d_in[11];
    const float* tp_g   = (const float*)d_in[12];
    const float* tp_be  = (const float*)d_in[13];
    const float* mg_w   = (const float*)d_in[14];
    const float* mg_b   = (const float*)d_in[15];
    const float* w0     = (const float*)d_in[16];
    const float* b0     = (const float*)d_in[17];
    const float* w1     = (const float*)d_in[18];
    const float* b1     = (const float*)d_in[19];
    float* out = (float*)d_out;

    // ---- carve workspace ---------------------------------------------------
    char* w = (char*)d_ws;
    size_t off = 0;
    auto carve = [&](size_t bytes) {
        char* p = w + off;
        off += (bytes + 255) & ~(size_t)255;
        return p;
    };
    const size_t nMH = (size_t)NM * NH;
    float*  hl_f  = (float*) carve(nMH * 4);
    bf16_t* hl_b  = (bf16_t*)carve(nMH * 2);
    bf16_t* x_b   = (bf16_t*)carve(nMH * 2);   // XS/XT, then reused for ctx
    bf16_t* q_b   = (bf16_t*)carve(nMH * 2);   // q, then ff t1
    bf16_t* kk_b  = (bf16_t*)carve(nMH * 2);   // k, then ff t2
    bf16_t* v_b   = (bf16_t*)carve(nMH * 2);
    bf16_t* ys_b  = (bf16_t*)carve(nMH * 2);
    bf16_t* yt_b  = (bf16_t*)carve(nMH * 2);
    float*  att   = (float*) carve((size_t)NB * NH * NI * NI * 4);
    float*  scr_f = (float*) carve(nMH * 4);   // ff out / merge acc / wd0 out
    float*  es_f  = (float*) carve((size_t)NI * NS * NH * 4);
    bf16_t* sp_wb = (bf16_t*)carve((size_t)NL * 6 * NH * NH * 2);
    bf16_t* tp_wb = (bf16_t*)carve((size_t)NL * 6 * NH * NH * 2);
    bf16_t* mg_wb = (bf16_t*)carve((size_t)NL * NH * 3 * NH * 2);
    bf16_t* w0b   = (bf16_t*)carve((size_t)NH * NH * 2);
    float*  part  = (float*) carve((size_t)NB * 128 * 4);
    (void)ws_size; (void)in_sizes; (void)n_in; (void)out_size;

    const dim3 gg(NM / 128, NH / 64);          // GEMM grid: (256, 4)
    auto gemm = [&](const bf16_t* A, const bf16_t* Wm, const float* bias,
                    const float* Cin, float* Cf, bf16_t* Cb2, int ldw, int relu) {
        k_gemm<<<gg, 256, 0, stream>>>(A, Wm, bias, Cin, Cf, Cb2, NH, NH, ldw, relu);
    };

    // ---- convert weights to bf16 once per call ----------------------------
    auto cvt = [&](const float* src, bf16_t* dst, int n) {
        k_f32_to_bf16<<<(n + 255) / 256, 256, 0, stream>>>(src, dst, n);
    };
    cvt(sp_w, sp_wb, NL * 6 * NH * NH);
    cvt(tp_w, tp_wb, NL * 6 * NH * NH);
    cvt(mg_w, mg_wb, NL * NH * 3 * NH);
    cvt(w0,   w0b,   NH * NH);

    // ---- hl = conv expand --------------------------------------------------
    k_conv<<<(NM * NH) / 256, 256, 0, stream>>>(x, cw, cb, hl_f, hl_b);

    // ---- attention block (shared between spatial & temporal) --------------
    auto attn = [&](const bf16_t* lwb, const float* lb,
                    const float* g_, const float* be_, bf16_t* Y) {
        gemm(x_b, lwb + 0 * NH * NH, lb + 0 * NH, nullptr, nullptr, q_b,  NH, 0);
        gemm(x_b, lwb + 1 * NH * NH, lb + 1 * NH, nullptr, nullptr, kk_b, NH, 0);
        gemm(x_b, lwb + 2 * NH * NH, lb + 2 * NH, nullptr, nullptr, v_b,  NH, 0);
        k_energy <<<4096, 256, 0, stream>>>(q_b, kk_b, att, 1.f / 16.f);
        k_softmax<<<(NB * NH * NI) / 256, 256, 0, stream>>>(att);
        k_context<<<4096, 256, 0, stream>>>(att, v_b, x_b);        // ctx -> x_b
        gemm(x_b,  lwb + 3 * NH * NH, lb + 3 * NH, nullptr, nullptr, q_b,  NH, 1);
        gemm(q_b,  lwb + 4 * NH * NH, lb + 4 * NH, nullptr, nullptr, kk_b, NH, 1);
        gemm(kk_b, lwb + 5 * NH * NH, lb + 5 * NH, nullptr, scr_f, nullptr, NH, 0);
        k_layernorm<<<NM / 8, 256, 0, stream>>>(scr_f, hl_f, g_, be_, Y);
    };

    for (int l = 0; l < NL; ++l) {
        // spatial branch: XS = hl + adj @ sp_was[l]
        k_es<<<(NI * NS * NH) / 256, 256, 0, stream>>>(
            adj, sp_was + (size_t)l * NI * NS * NH, es_f);
        k_add_bf16<<<(NM * NH) / 256, 256, 0, stream>>>(
            hl_f, es_f, x_b, (unsigned)(NI * NS * NH - 1));
        attn(sp_wb + (size_t)l * 6 * NH * NH, sp_b + l * 6 * NH,
             sp_g + l * NH, sp_be + l * NH, ys_b);

        // temporal branch: XT = hl + pos[l]
        k_add_bf16<<<(NM * NH) / 256, 256, 0, stream>>>(
            hl_f, tp_pos + (size_t)l * NS * NH, x_b, (unsigned)(NS * NH - 1));
        attn(tp_wb + (size_t)l * 6 * NH * NH, tp_b + l * 6 * NH,
             tp_g + l * NH, tp_be + l * NH, yt_b);

        // merge: hl' = relu([hl|YS|YT] @ mg_w[l]^T + mg_b[l]) as 3 K=256 GEMMs
        const bf16_t* mw = mg_wb + (size_t)l * NH * 3 * NH;
        gemm(hl_b, mw + 0 * NH, mg_b + l * NH, nullptr, scr_f, nullptr, 3 * NH, 0);
        gemm(ys_b, mw + 1 * NH, nullptr,       scr_f,   scr_f, nullptr, 3 * NH, 0);
        gemm(yt_b, mw + 2 * NH, nullptr,       scr_f,   hl_f,  hl_b,    3 * NH, 1);
    }

    // ---- head: h = hl @ wd0^T + b0 ; logits = <h, wd1> + b1 ; sigmoid -----
    gemm(hl_b, w0b, b0, nullptr, scr_f, nullptr, NH, 0);
    k_dot_partial<<<NB * 128, 256, 0, stream>>>(scr_f, w1, part);
    k_dot_final  <<<NB, 128, 0, stream>>>(part, b1, out);
}
